// SelfAttention_3856880632314
// MI455X (gfx1250) — matrix operand order repaired and checked by reading, not compiled, not run on hardware
//
#include <hip/hip_runtime.h>
#include <hip/hip_bf16.h>
#include <math.h>
#include <stdint.h>

// ---------------------------------------------------------------------------
// Self-attention block (GroupNorm -> QKV -> MHA softmax -> out-proj + residual)
// B=8, C=512, H=W=32 (P=1024 pixels), 8 heads x 64 dim, GROUPS=32.
// GEMMs on v_wmma_f32_16x16x32_f16; projection GEMMs stage A/B tiles in LDS
// via global_load_async_to_lds_b128 (ASYNCcnt), double-buffered with a runtime
// LDS offset toggle + nounroll so accumulators stay pinned (no C->D movs).
// ---------------------------------------------------------------------------

typedef _Float16 v8h  __attribute__((ext_vector_type(8)));
typedef _Float16 v16h __attribute__((ext_vector_type(16)));
typedef float    v8f  __attribute__((ext_vector_type(8)));

#define BATCH   8
#define CH      512
#define PIX     1024
#define NHEAD   8
#define HDIM    64
#define OC3     1536
#define BUFH    4096                    // 128*32 halves per LDS buffer

// ---- WMMA fragment loaders (layouts per CDNA5 ISA 7.12.2) ------------------
// A (16x32 f16, row-major [m][k]): lane m=lane&15; halves 0..7 = K kb..kb+7,
// halves 8..15 = K kb+16..kb+23, kb = 8*(lane>>4).
__device__ __forceinline__ v16h load_a_frag(const _Float16* a, int stride,
                                            int k0, int lane) {
    const int m  = lane & 15;
    const int hb = lane >> 4;
    const _Float16* p = a + (size_t)m * stride + k0 + hb * 8;
    v8h lo = *(const v8h*)(p);
    v8h hi = *(const v8h*)(p + 16);
    return __builtin_shufflevector(lo, hi, 0,1,2,3,4,5,6,7,8,9,10,11,12,13,14,15);
}

// B (32x16 f16) given as B^T row-major [n][k]: lane n=lane&15; halves 0..7 =
// K kb..kb+7, halves 8..15 = K kb+16..kb+23, kb = 8*(lane>>4).
__device__ __forceinline__ v16h load_b_frag(const _Float16* bt, int stride,
                                            int k0, int lane) {
    const int n  = lane & 15;
    const int hb = lane >> 4;
    const _Float16* p = bt + (size_t)n * stride + k0 + hb * 8;
    v8h lo = *(const v8h*)(p);
    v8h hi = *(const v8h*)(p + 16);
    return __builtin_shufflevector(lo, hi, 0,1,2,3,4,5,6,7,8,9,10,11,12,13,14,15);
}

__device__ __forceinline__ v8f wmma16(v16h a, v16h b, v8f c) {
    return __builtin_amdgcn_wmma_f32_16x16x32_f16(false, a, false, b,
                                                  (short)0, c, false, false);
}

#define V8F_ZERO {0.f,0.f,0.f,0.f,0.f,0.f,0.f,0.f}

// ---- CDNA5 async global->LDS copy (16B per lane, ASYNCcnt-tracked) ---------
// LDS destination address = low 32 bits of generic shared pointer;
// global address in a VGPR pair (GV mode).
__device__ __forceinline__ void async_copy_b128(void* lds_ptr, const void* gptr) {
    uint32_t lds_off = (uint32_t)(uintptr_t)lds_ptr;
    uint64_t gaddr   = (uint64_t)(uintptr_t)gptr;
    asm volatile("global_load_async_to_lds_b128 %0, %1, off"
                 :: "v"(lds_off), "v"(gaddr) : "memory");
}
__device__ __forceinline__ void wait_async_0() {
    asm volatile("s_wait_asynccnt 0x0" ::: "memory");
}

// ---- K0: fp32 -> f16 weight conversion -------------------------------------
__global__ void cvt_f16_kernel(const float* __restrict__ src,
                               _Float16* __restrict__ dst, int n) {
    int i = blockIdx.x * 256 + threadIdx.x;
    if (i < n) dst[i] = (_Float16)src[i];
}

// ---- K1: GroupNorm; writes xn transposed [b][p][c] as f16 ------------------
__global__ __launch_bounds__(256)
void gn_kernel(const float* __restrict__ x, const float* __restrict__ gw,
               const float* __restrict__ gb, _Float16* __restrict__ xnT) {
    __shared__ float s1[256];
    __shared__ float s2[256];
    const int b  = blockIdx.x >> 5;           // 32 groups
    const int g  = blockIdx.x & 31;
    const int c0 = g * 16;
    const int tid = threadIdx.x;
    const float* xb = x + ((size_t)b * CH + c0) * PIX;  // 16*1024 contiguous

    float sum = 0.f, sq = 0.f;
    #pragma unroll 4
    for (int i = 0; i < 64; ++i) {
        float v = xb[tid + i * 256];
        sum += v; sq += v * v;
    }
    s1[tid] = sum; s2[tid] = sq;
    __syncthreads();
    for (int s = 128; s > 0; s >>= 1) {
        if (tid < s) { s1[tid] += s1[tid + s]; s2[tid] += s2[tid + s]; }
        __syncthreads();
    }
    const float mean = s1[0] * (1.f / 16384.f);
    const float var  = s2[0] * (1.f / 16384.f) - mean * mean;
    const float rstd = rsqrtf(var + 1e-5f);

    #pragma unroll 4
    for (int i = 0; i < 64; ++i) {
        int lin = tid + i * 256;
        int cc  = lin >> 10;
        int p   = lin & 1023;
        float v = (xb[lin] - mean) * rstd * gw[c0 + cc] + gb[c0 + cc];
        xnT[((size_t)b * PIX + p) * CH + c0 + cc] = (_Float16)v;
    }
}

// ---- K2: QKV projection, LDS-tiled 128x128 with async double buffering. ----
// qkv[o,p] = sum_c W[o,c]*xnT[p,c] + bias[o].  q scaled by 1/sqrt(64).
// 8 waves (4x2): each wave computes 32(M) x 64(N).
__global__ __launch_bounds__(256)
void qkv_kernel(const _Float16* __restrict__ W, const float* __restrict__ bias,
                const _Float16* __restrict__ xnT,
                _Float16* __restrict__ qT, _Float16* __restrict__ kT,
                _Float16* __restrict__ vB) {
    __shared__ _Float16 sA[2 * BUFH];       // A k-strips (o rows), 2 buffers
    __shared__ _Float16 sB[2 * BUFH];       // B^T k-strips (p rows), 2 buffers

    const int tid  = threadIdx.x;
    const int lane = tid & 31;
    const int w    = tid >> 5;
    int blk = blockIdx.x;
    const int pt = blk & 7;   blk >>= 3;    // 1024/128 = 8
    const int ot = blk % 12;                // 1536/128 = 12
    const int b  = blk / 12;
    const int o0 = ot * 128, p0 = pt * 128;

    const _Float16* Ag = W + (size_t)o0 * CH;
    const _Float16* Bg = xnT + ((size_t)b * PIX + p0) * CH;

    // chunk mapping: 512 16B-chunks per strip, 2 per thread
    const int c0i = tid * 32 + (tid & 3) * 8 - (tid & 3) * 32; // row0*128+q0
    const int ofs0 = (tid >> 2) * 32 + (tid & 3) * 8;
    const int ofs1 = ((tid + 256) >> 2) * 32 + ((tid + 256) & 3) * 8;
    const size_t g0 = (size_t)(tid >> 2) * CH + (tid & 3) * 8;
    const size_t g1 = (size_t)((tid + 256) >> 2) * CH + ((tid + 256) & 3) * 8;
    (void)c0i;

    const int wm = w >> 1, wn = w & 1;      // 4 x 2 wave grid
    const int mb = wm * 32, nb = wn * 64;

    v8f c00 = V8F_ZERO, c01 = V8F_ZERO, c02 = V8F_ZERO, c03 = V8F_ZERO;
    v8f c10 = V8F_ZERO, c11 = V8F_ZERO, c12 = V8F_ZERO, c13 = V8F_ZERO;

    auto stage = [&](int bufo, int kk) {
        async_copy_b128(&sA[bufo + ofs0], Ag + g0 + kk);
        async_copy_b128(&sA[bufo + ofs1], Ag + g1 + kk);
        async_copy_b128(&sB[bufo + ofs0], Bg + g0 + kk);
        async_copy_b128(&sB[bufo + ofs1], Bg + g1 + kk);
    };
    auto compute = [&](int bufo) {
        const _Float16* sa = &sA[bufo];
        const _Float16* sb = &sB[bufo];
        v16h a0 = load_a_frag(sa + (size_t)(mb) * 32, 32, 0, lane);
        v16h a1 = load_a_frag(sa + (size_t)(mb + 16) * 32, 32, 0, lane);
        v16h b0 = load_b_frag(sb + (size_t)(nb) * 32, 32, 0, lane);
        v16h b1 = load_b_frag(sb + (size_t)(nb + 16) * 32, 32, 0, lane);
        v16h b2 = load_b_frag(sb + (size_t)(nb + 32) * 32, 32, 0, lane);
        v16h b3 = load_b_frag(sb + (size_t)(nb + 48) * 32, 32, 0, lane);
        c00 = wmma16(a0, b0, c00);  c10 = wmma16(a1, b0, c10);
        c01 = wmma16(a0, b1, c01);  c11 = wmma16(a1, b1, c11);
        c02 = wmma16(a0, b2, c02);  c12 = wmma16(a1, b2, c12);
        c03 = wmma16(a0, b3, c03);  c13 = wmma16(a1, b3, c13);
    };

    stage(0, 0);                            // prologue
    int cur = 0;
    #pragma clang loop unroll(disable)
    for (int ks = 0; ks < 15; ++ks) {       // uniform main loop, K = 512/32
        wait_async_0();                     // this wave's strip landed
        __syncthreads();                    // everyone's strip landed / buffer free
        stage(cur ^ BUFH, (ks + 1) * 32);   // prefetch next (overlaps compute)
        compute(cur);
        cur ^= BUFH;
    }
    wait_async_0();                         // peeled last step
    __syncthreads();
    compute(cur);

    v8f acc[2][4] = {{c00, c01, c02, c03}, {c10, c11, c12, c13}};
    const int hb = lane >> 4, n = lane & 15;
    #pragma unroll
    for (int tt = 0; tt < 2; ++tt) {
        #pragma unroll
        for (int nt = 0; nt < 4; ++nt) {
            #pragma unroll
            for (int r = 0; r < 8; ++r) {
                const int o = o0 + mb + tt * 16 + r + 8 * hb;
                const int p = p0 + nb + nt * 16 + n;
                const float v = acc[tt][nt][r] + bias[o];
                if (o < CH) {               // q (pre-scaled by 1/8)
                    int head = o >> 6, d = o & 63;
                    qT[(((size_t)b * NHEAD + head) * PIX + p) * HDIM + d] =
                        (_Float16)(v * 0.125f);
                } else if (o < 2 * CH) {    // k
                    int o2 = o - CH, head = o2 >> 6, d = o2 & 63;
                    kT[(((size_t)b * NHEAD + head) * PIX + p) * HDIM + d] = (_Float16)v;
                } else {                    // v
                    int o2 = o - 2 * CH, head = o2 >> 6, d = o2 & 63;
                    vB[(((size_t)b * NHEAD + head) * HDIM + d) * PIX + p] = (_Float16)v;
                }
            }
        }
    }
}

// ---- K3: fused scores + softmax + attn@v for one 16-row strip --------------
// Block = 256 threads (8 waves). Wave w owns columns [w*128, w*128+128).
__global__ __launch_bounds__(256)
void attn_kernel(const _Float16* __restrict__ qT, const _Float16* __restrict__ kT,
                 const _Float16* __restrict__ vB, _Float16* __restrict__ aT) {
    __shared__ _Float16 sc[16 * PIX];       // f16 probabilities, 32 KB
    __shared__ float red[8][16];            // cross-wave row stats
    __shared__ float obuf[16 * HDIM];       // output reduction, 4 KB

    const int tid  = threadIdx.x;
    const int lane = tid & 31;
    const int w    = tid >> 5;
    const int rt   = blockIdx.x & 63;       // row tile within P
    const int bh   = blockIdx.x >> 6;       // b*8 + head

    const _Float16* q = qT + ((size_t)bh * PIX + rt * 16) * HDIM;
    const _Float16* k = kT + (size_t)bh * PIX * HDIM;
    const _Float16* v = vB + (size_t)bh * HDIM * PIX;

    for (int i = tid; i < 16 * HDIM; i += 256) obuf[i] = 0.f;

    const int cbase = w * 128;
    const int hb = lane >> 4, n = lane & 15;

    // --- scores S[16][128] for this wave (q pre-scaled by 1/8) ---
    v16h aq0 = load_a_frag(q, HDIM, 0, lane);
    v16h aq1 = load_a_frag(q, HDIM, 32, lane);
    v8f s[8];
    #pragma unroll
    for (int j = 0; j < 8; ++j) {
        const _Float16* kb = k + (size_t)(cbase + j * 16) * HDIM;
        v16h b0 = load_b_frag(kb, HDIM, 0, lane);
        v16h b1 = load_b_frag(kb, HDIM, 32, lane);
        v8f t = V8F_ZERO;
        t = wmma16(aq0, b0, t);
        t = wmma16(aq1, b1, t);
        s[j] = t;
    }

    // --- row max (lanes within half -> across waves via LDS) ---
    float lm[8];
    #pragma unroll
    for (int r = 0; r < 8; ++r) {
        float m = s[0][r];
        #pragma unroll
        for (int j = 1; j < 8; ++j) m = fmaxf(m, s[j][r]);
        lm[r] = m;
    }
    #pragma unroll
    for (int r = 0; r < 8; ++r) {
        #pragma unroll
        for (int mk = 1; mk < 16; mk <<= 1)
            lm[r] = fmaxf(lm[r], __shfl_xor(lm[r], mk, 32));
    }
    if (n == 0) {
        #pragma unroll
        for (int r = 0; r < 8; ++r) red[w][hb * 8 + r] = lm[r];
    }
    __syncthreads();
    float gmax[8];
    #pragma unroll
    for (int r = 0; r < 8; ++r) {
        float m = red[0][hb * 8 + r];
        #pragma unroll
        for (int ww = 1; ww < 8; ++ww) m = fmaxf(m, red[ww][hb * 8 + r]);
        gmax[r] = m;
    }
    __syncthreads();

    // --- exp + row sum ---
    float ls[8];
    #pragma unroll
    for (int r = 0; r < 8; ++r) ls[r] = 0.f;
    #pragma unroll
    for (int j = 0; j < 8; ++j) {
        #pragma unroll
        for (int r = 0; r < 8; ++r) {
            float e = __expf(s[j][r] - gmax[r]);
            s[j][r] = e;
            ls[r] += e;
        }
    }
    #pragma unroll
    for (int r = 0; r < 8; ++r) {
        #pragma unroll
        for (int mk = 1; mk < 16; mk <<= 1)
            ls[r] += __shfl_xor(ls[r], mk, 32);
    }
    if (n == 0) {
        #pragma unroll
        for (int r = 0; r < 8; ++r) red[w][hb * 8 + r] = ls[r];
    }
    __syncthreads();
    float rinv[8];
    #pragma unroll
    for (int r = 0; r < 8; ++r) {
        float sm = 0.f;
        #pragma unroll
        for (int ww = 0; ww < 8; ++ww) sm += red[ww][hb * 8 + r];
        rinv[r] = 1.f / sm;
    }

    // --- normalized f16 probabilities to LDS (A-layout source) ---
    #pragma unroll
    for (int j = 0; j < 8; ++j) {
        #pragma unroll
        for (int r = 0; r < 8; ++r) {
            int row = r + 8 * hb;
            int col = cbase + j * 16 + n;
            sc[row * PIX + col] = (_Float16)(s[j][r] * rinv[r]);
        }
    }
    __syncthreads();

    // --- O += S[:, wave-chunk] @ v^T  (K split across waves) ---
    v8f o[4] = {V8F_ZERO, V8F_ZERO, V8F_ZERO, V8F_ZERO};
    #pragma unroll
    for (int kk = 0; kk < 4; ++kk) {
        const int kg = cbase + kk * 32;
        v16h af = load_a_frag(sc, PIX, kg, lane);
        #pragma unroll
        for (int nt = 0; nt < 4; ++nt) {
            v16h bf = load_b_frag(v + (size_t)(nt * 16) * PIX, PIX, kg, lane);
            o[nt] = wmma16(af, bf, o[nt]);
        }
    }
    #pragma unroll
    for (int nt = 0; nt < 4; ++nt) {
        #pragma unroll
        for (int r = 0; r < 8; ++r) {
            int row = r + 8 * hb;
            int d   = nt * 16 + n;
            atomicAdd(&obuf[row * HDIM + d], o[nt][r]);
        }
    }
    __syncthreads();

    // --- write 16x64 strip to aT[b][p][head*64+d] (f16) ---
    const int b = bh >> 3, head = bh & 7;
    for (int i = tid; i < 16 * HDIM; i += 256) {
        int row = i >> 6, d = i & 63;
        aT[((size_t)b * PIX + rt * 16 + row) * CH + head * HDIM + d] =
            (_Float16)obuf[i];
    }
}

// ---- K4: out projection + bias + residual, LDS-tiled like K2 ---------------
__global__ __launch_bounds__(256)
void proj_kernel(const _Float16* __restrict__ W, const float* __restrict__ bias,
                 const _Float16* __restrict__ aT, const float* __restrict__ x,
                 float* __restrict__ out) {
    __shared__ _Float16 sA[2 * BUFH];
    __shared__ _Float16 sB[2 * BUFH];

    const int tid  = threadIdx.x;
    const int lane = tid & 31;
    const int w    = tid >> 5;
    int blk = blockIdx.x;
    const int pt = blk & 7;   blk >>= 3;    // 1024/128 = 8
    const int ot = blk % 4;                 // 512/128 = 4
    const int b  = blk / 4;
    const int o0 = ot * 128, p0 = pt * 128;

    const _Float16* Ag = W + (size_t)o0 * CH;
    const _Float16* Bg = aT + ((size_t)b * PIX + p0) * CH;

    const int ofs0 = (tid >> 2) * 32 + (tid & 3) * 8;
    const int ofs1 = ((tid + 256) >> 2) * 32 + ((tid + 256) & 3) * 8;
    const size_t g0 = (size_t)(tid >> 2) * CH + (tid & 3) * 8;
    const size_t g1 = (size_t)((tid + 256) >> 2) * CH + ((tid + 256) & 3) * 8;

    const int wm = w >> 1, wn = w & 1;
    const int mb = wm * 32, nb = wn * 64;

    v8f c00 = V8F_ZERO, c01 = V8F_ZERO, c02 = V8F_ZERO, c03 = V8F_ZERO;
    v8f c10 = V8F_ZERO, c11 = V8F_ZERO, c12 = V8F_ZERO, c13 = V8F_ZERO;

    auto stage = [&](int bufo, int kk) {
        async_copy_b128(&sA[bufo + ofs0], Ag + g0 + kk);
        async_copy_b128(&sA[bufo + ofs1], Ag + g1 + kk);
        async_copy_b128(&sB[bufo + ofs0], Bg + g0 + kk);
        async_copy_b128(&sB[bufo + ofs1], Bg + g1 + kk);
    };
    auto compute = [&](int bufo) {
        const _Float16* sa = &sA[bufo];
        const _Float16* sb = &sB[bufo];
        v16h a0 = load_a_frag(sa + (size_t)(mb) * 32, 32, 0, lane);
        v16h a1 = load_a_frag(sa + (size_t)(mb + 16) * 32, 32, 0, lane);
        v16h b0 = load_b_frag(sb + (size_t)(nb) * 32, 32, 0, lane);
        v16h b1 = load_b_frag(sb + (size_t)(nb + 16) * 32, 32, 0, lane);
        v16h b2 = load_b_frag(sb + (size_t)(nb + 32) * 32, 32, 0, lane);
        v16h b3 = load_b_frag(sb + (size_t)(nb + 48) * 32, 32, 0, lane);
        c00 = wmma16(a0, b0, c00);  c10 = wmma16(a1, b0, c10);
        c01 = wmma16(a0, b1, c01);  c11 = wmma16(a1, b1, c11);
        c02 = wmma16(a0, b2, c02);  c12 = wmma16(a1, b2, c12);
        c03 = wmma16(a0, b3, c03);  c13 = wmma16(a1, b3, c13);
    };

    stage(0, 0);
    int cur = 0;
    #pragma clang loop unroll(disable)
    for (int ks = 0; ks < 15; ++ks) {
        wait_async_0();
        __syncthreads();
        stage(cur ^ BUFH, (ks + 1) * 32);
        compute(cur);
        cur ^= BUFH;
    }
    wait_async_0();
    __syncthreads();
    compute(cur);

    v8f acc[2][4] = {{c00, c01, c02, c03}, {c10, c11, c12, c13}};
    const int hb = lane >> 4, n = lane & 15;
    #pragma unroll
    for (int tt = 0; tt < 2; ++tt) {
        #pragma unroll
        for (int nt = 0; nt < 4; ++nt) {
            #pragma unroll
            for (int r = 0; r < 8; ++r) {
                const int o = o0 + mb + tt * 16 + r + 8 * hb;
                const int p = p0 + nb + nt * 16 + n;
                const size_t idx = ((size_t)b * CH + o) * PIX + p;
                out[idx] = acc[tt][nt][r] + bias[o] + x[idx];
            }
        }
    }
}

// ---------------------------------------------------------------------------
extern "C" void kernel_launch(void* const* d_in, const int* in_sizes, int n_in,
                              void* d_out, int out_size, void* d_ws, size_t ws_size,
                              hipStream_t stream) {
    const float* x       = (const float*)d_in[0];
    const float* gnw     = (const float*)d_in[1];
    const float* gnb     = (const float*)d_in[2];
    const float* qkvW32  = (const float*)d_in[3];
    const float* qkvBias = (const float*)d_in[4];
    const float* outW32  = (const float*)d_in[5];
    const float* outBias = (const float*)d_in[6];
    float* out = (float*)d_out;

    // workspace carve-up (f16 buffers)
    char* ws = (char*)d_ws;
    size_t off = 0;
    _Float16* qkvW16 = (_Float16*)(ws + off); off += (size_t)OC3 * CH * 2;
    _Float16* outW16 = (_Float16*)(ws + off); off += (size_t)CH * CH * 2;
    _Float16* xnT    = (_Float16*)(ws + off); off += (size_t)BATCH * PIX * CH * 2;
    _Float16* qT     = (_Float16*)(ws + off); off += (size_t)BATCH * NHEAD * PIX * HDIM * 2;
    _Float16* kT     = (_Float16*)(ws + off); off += (size_t)BATCH * NHEAD * PIX * HDIM * 2;
    _Float16* vB     = (_Float16*)(ws + off); off += (size_t)BATCH * NHEAD * HDIM * PIX * 2;
    _Float16* aT     = (_Float16*)(ws + off); off += (size_t)BATCH * PIX * CH * 2;
    (void)ws_size; (void)in_sizes; (void)n_in; (void)out_size;

    // K0: weight conversion
    cvt_f16_kernel<<<(OC3 * CH + 255) / 256, 256, 0, stream>>>(qkvW32, qkvW16, OC3 * CH);
    cvt_f16_kernel<<<(CH * CH + 255) / 256, 256, 0, stream>>>(outW32, outW16, CH * CH);

    // K1: GroupNorm (B*GROUPS blocks)
    gn_kernel<<<BATCH * 32, 256, 0, stream>>>(x, gnw, gnb, xnT);

    // K2: QKV GEMM — 128x128 LDS-tiled: B * (1536/128) * (1024/128) blocks
    qkv_kernel<<<BATCH * 12 * 8, 256, 0, stream>>>(qkvW16, qkvBias, xnT, qT, kT, vB);

    // K3: fused attention — (B*NHEAD) * (1024/16) strips, 8 waves each
    attn_kernel<<<BATCH * NHEAD * 64, 256, 0, stream>>>(qT, kT, vB, aT);

    // K4: output projection + residual — B * (512/128) * (1024/128) tiles
    proj_kernel<<<BATCH * 4 * 8, 256, 0, stream>>>(outW16, outBias, aT, x, out);
}
